// GroupedQueryAttention_82884278878229
// MI455X (gfx1250) — matrix operand order repaired and checked by reading, not compiled, not run on hardware
//
#include <hip/hip_runtime.h>
#include <cstdint>
#include <cstddef>

typedef __attribute__((ext_vector_type(16))) __bf16 bf16x16;
typedef __attribute__((ext_vector_type(8)))  float  f32x8;

namespace {
constexpr int BATCH = 2;
constexpr int SEQ   = 2048;
constexpr int CH    = 2048;          // model dim
constexpr int NH    = 32;            // query heads
constexpr int NKV   = 8;             // kv heads
constexpr int HD    = 64;            // head dim
constexpr int GQ    = NH / NKV;      // 4
constexpr int MR    = BATCH * SEQ;   // 4096 token rows
constexpr int QN    = NH * HD;       // 2048
constexpr int KVN   = 2 * NKV * HD;  // 1024 (k then v halves)
}

// ds_swizzle_b32 XOR butterfly within each 16-lane group (and_mask=0x1f keeps
// lanes in-wave; xor masks <=8 keep lanes within their 16-lane half).
#define SWZF(x, m)                                                          \
  __builtin_bit_cast(float, __builtin_amdgcn_ds_swizzle(                     \
                                __builtin_bit_cast(int, (x)),                \
                                (((m) << 10) | 0x1f)))

// ---- async global->LDS copies (GLOBAL_LOAD_ASYNC_TO_LDS_B128, ASYNCcnt) ----
#if defined(__gfx1250__) && \
    __has_builtin(__builtin_amdgcn_global_load_async_to_lds_b128)
#define USE_ASYNC_LDS 1
typedef int v4i __attribute__((__vector_size__(16)));
typedef __attribute__((address_space(1))) v4i gv4i;   // global int4
typedef __attribute__((address_space(3))) v4i lv4i;   // LDS int4
static __device__ __forceinline__ void async_cp16(const void* g, void* l) {
  // LDS generic pointers carry the LDS byte offset in their low 32 bits.
  __builtin_amdgcn_global_load_async_to_lds_b128(
      (gv4i*)(uintptr_t)g, (lv4i*)(uint32_t)(uintptr_t)l, 0, 0);
}
static __device__ __forceinline__ void async_wait0() {
#if __has_builtin(__builtin_amdgcn_s_wait_asynccnt)
  __builtin_amdgcn_s_wait_asynccnt(0);
#else
  asm volatile("s_wait_asynccnt 0" ::: "memory");
#endif
}
#else
#define USE_ASYNC_LDS 0
#endif

// wave-local LDS fence: lanes of one wave exchanging through wave-private LDS
// only need the wave's own DScnt drained (LDS ops are in-order per wave).
static __device__ __forceinline__ void lds_wave_fence() {
#if defined(__gfx1250__)
  asm volatile("s_wait_dscnt 0" ::: "memory");
#else
  __syncthreads();
#endif
}

union FragU { uint4 u4[2]; bf16x16 v; };

static __device__ __forceinline__ bf16x16 ldfrag(const __bf16* p0, const __bf16* p1) {
  FragU f;
  f.u4[0] = *(const uint4*)p0;
  f.u4[1] = *(const uint4*)p1;
  return f.v;
}

static __device__ __forceinline__ f32x8 wmma_bf16(bf16x16 a, bf16x16 b, f32x8 c) {
  // D = A(16x32 bf16) * B(32x16 bf16) + C(16x16 f32)
  return __builtin_amdgcn_wmma_f32_16x16x32_bf16(false, a, false, b, (short)0, c,
                                                 false, false);
}

// ---------------------------------------------------------------- fp32 -> bf16
__global__ void cvt_f32_to_bf16(const float* __restrict__ in,
                                __bf16* __restrict__ out, int n4) {
  int i = blockIdx.x * blockDim.x + threadIdx.x;
  if (i >= n4) return;
  float4 v = ((const float4*)in)[i];
  union { __bf16 b[4]; uint2 u; } o;
  o.b[0] = (__bf16)v.x; o.b[1] = (__bf16)v.y;
  o.b[2] = (__bf16)v.z; o.b[3] = (__bf16)v.w;
  ((uint2*)out)[i] = o.u;
}

// ------------------------------------------------------- C = A[M,K] * W[N,K]^T
// block tile 128x128, 8 waves (4 along M x 2 along N), each wave 32x64,
// K-step 64 -> 16 v_wmma_f32_16x16x32_bf16 per wave per K-step.
template <bool OUT_BF16>
__global__ void __launch_bounds__(256)
gemm_xwT(const __bf16* __restrict__ A, const __bf16* __restrict__ W,
         void* __restrict__ Cout, int Kdim, int Ndim, float outScale) {
  constexpr int LDT = 72;                       // padded LDS row stride (elems)
  __shared__ __bf16 sA[128 * LDT];
  __shared__ __bf16 sW[128 * LDT];

  const int tid  = threadIdx.x;
  const int wave = tid >> 5, lane = tid & 31;
  const int hl   = lane >> 4, ln = lane & 15;
  const int m0   = blockIdx.y * 128, n0 = blockIdx.x * 128;
  const int wm   = (wave & 3) * 32;             // wave M offset in tile
  const int wn   = (wave >> 2) * 64;            // wave N offset in tile

  f32x8 acc[2][4];
#pragma unroll
  for (int ms = 0; ms < 2; ++ms)
#pragma unroll
    for (int ns = 0; ns < 4; ++ns)
#pragma unroll
      for (int r = 0; r < 8; ++r) acc[ms][ns][r] = 0.0f;

  for (int k0 = 0; k0 < Kdim; k0 += 64) {
    __syncthreads();
    // cooperative load: 128 rows x 64 cols each for A and W, 16B chunks
#pragma unroll
    for (int j = 0; j < 4; ++j) {
      int c = tid * 4 + j;                      // 0..1023
      int row = c >> 3, cc = (c & 7) * 8;
      const size_t ga = (size_t)(m0 + row) * Kdim + k0 + cc;
      const size_t gw = (size_t)(n0 + row) * Kdim + k0 + cc;
#if USE_ASYNC_LDS
      async_cp16(&A[ga], &sA[row * LDT + cc]);
      async_cp16(&W[gw], &sW[row * LDT + cc]);
#else
      *(uint4*)&sA[row * LDT + cc] = *(const uint4*)&A[ga];
      *(uint4*)&sW[row * LDT + cc] = *(const uint4*)&W[gw];
#endif
      __builtin_prefetch(&A[ga + 64], 0, 1);    // next K-step (speculative)
      __builtin_prefetch(&W[gw + 64], 0, 1);
    }
#if USE_ASYNC_LDS
    async_wait0();
#endif
    __syncthreads();

    bf16x16 af[2][2], wf[4][2];
#pragma unroll
    for (int ms = 0; ms < 2; ++ms)
#pragma unroll
      for (int j = 0; j < 2; ++j) {
        const __bf16* base = &sA[(wm + ms * 16 + ln) * LDT + j * 32];
        af[ms][j] = ldfrag(base + 8 * hl, base + 16 + 8 * hl);
      }
#pragma unroll
    for (int ns = 0; ns < 4; ++ns)
#pragma unroll
      for (int j = 0; j < 2; ++j) {
        const __bf16* base = &sW[(wn + ns * 16 + ln) * LDT + j * 32];
        wf[ns][j] = ldfrag(base + 8 * hl, base + 16 + 8 * hl);
      }
#pragma unroll
    for (int ms = 0; ms < 2; ++ms)
#pragma unroll
      for (int ns = 0; ns < 4; ++ns) {
        acc[ms][ns] = wmma_bf16(af[ms][0], wf[ns][0], acc[ms][ns]);
        acc[ms][ns] = wmma_bf16(af[ms][1], wf[ns][1], acc[ms][ns]);
      }
  }

#pragma unroll
  for (int ms = 0; ms < 2; ++ms)
#pragma unroll
    for (int ns = 0; ns < 4; ++ns)
#pragma unroll
      for (int r = 0; r < 8; ++r) {
        const int row = m0 + wm + ms * 16 + r + 8 * hl;
        const int col = n0 + wn + ns * 16 + ln;
        const float v = acc[ms][ns][r] * outScale;
        if (OUT_BF16)
          ((__bf16*)Cout)[(size_t)row * Ndim + col] = (__bf16)v;
        else
          ((float*)Cout)[(size_t)row * Ndim + col] = v;
      }
}

// --------------------------------------------------------------- flash attention
// grid = (SEQ/64, BATCH*NH); 128 threads = 4 waves; each wave owns 16 query
// rows; key tiles of 32; online softmax in fp32; causal mask.
// Q is pre-scaled by 1/sqrt(D) in its projection GEMM.
__global__ void __launch_bounds__(128)
flash_attn(const __bf16* __restrict__ Qm, const __bf16* __restrict__ KVm,
           __bf16* __restrict__ Om) {
  constexpr int LQ = 72, LK = 72, LV = 40, LP = 40;
  __shared__ __bf16 sQ[64 * LQ];    // 64 q rows x 64 d
  __shared__ __bf16 sK[32 * LK];    // 32 keys x 64 d (row = key)
  __shared__ __bf16 sVt[64 * LV];   // transposed: row = d (64), col = key (32)
  __shared__ __bf16 sP[4 * 16 * LP];// per-wave P tile 16x32

  const int tid = threadIdx.x;
  const int w = tid >> 5, lane = tid & 31;
  const int hl = lane >> 4, ln = lane & 15;
  const int qt = blockIdx.x, bh = blockIdx.y;
  const int b = bh / NH, h = bh % NH, hk = h / GQ;
  const int q0 = qt * 64;
  const size_t qrow0 = (size_t)(b * SEQ + q0);

  // stage Q tile (64x64) and build per-wave A fragments (held in VGPRs)
#pragma unroll
  for (int j = 0; j < 4; ++j) {
    int c = tid * 4 + j;                        // 0..511
    int row = c >> 3, cc = (c & 7) * 8;
#if USE_ASYNC_LDS
    async_cp16(&Qm[(qrow0 + row) * QN + h * HD + cc], &sQ[row * LQ + cc]);
#else
    *(uint4*)&sQ[row * LQ + cc] =
        *(const uint4*)&Qm[(qrow0 + row) * QN + h * HD + cc];
#endif
  }
#if USE_ASYNC_LDS
  async_wait0();
#endif
  __syncthreads();
  bf16x16 qf[2];
#pragma unroll
  for (int j = 0; j < 2; ++j) {
    const __bf16* base = &sQ[(w * 16 + ln) * LQ + j * 32];
    qf[j] = ldfrag(base + 8 * hl, base + 16 + 8 * hl);
  }

  f32x8 o[4];
  float mrun[8], lpart[8];
#pragma unroll
  for (int nt = 0; nt < 4; ++nt)
#pragma unroll
    for (int r = 0; r < 8; ++r) o[nt][r] = 0.0f;
#pragma unroll
  for (int r = 0; r < 8; ++r) { mrun[r] = -3.4e38f; lpart[r] = 0.0f; }

  const int mbase = q0 + w * 16 + 8 * hl;        // query row of acc element r
  const int nkt = qt * 2 + 2;                    // causal: keys [0, q0+64)

  for (int kt = 0; kt < nkt; ++kt) {
    const int sb = kt * 32;
    __syncthreads();                             // protect LDS reuse
#pragma unroll
    for (int j = 0; j < 2; ++j) {
      int c = tid * 2 + j;                       // 0..255
      int row = c >> 3, cc = (c & 7) * 8;        // row = key, cc = d chunk
      const size_t g = (size_t)(b * SEQ + sb + row) * KVN + hk * HD;
#if USE_ASYNC_LDS
      async_cp16(&KVm[g + cc], &sK[row * LK + cc]);
#else
      *(uint4*)&sK[row * LK + cc] = *(const uint4*)&KVm[g + cc];
#endif
      union { uint4 u; __bf16 e[8]; } vv;        // V: transpose into sVt
      vv.u = *(const uint4*)&KVm[g + NKV * HD + cc];
#pragma unroll
      for (int t = 0; t < 8; ++t) sVt[(cc + t) * LV + row] = vv.e[t];
      // prefetch next key tile (speculative; dropped if OOB)
      __builtin_prefetch(&KVm[g + (size_t)32 * KVN + cc], 0, 1);
      __builtin_prefetch(&KVm[g + (size_t)32 * KVN + NKV * HD + cc], 0, 1);
    }
#if USE_ASYNC_LDS
    async_wait0();
#endif
    __syncthreads();

    // S = Q * K^T  (16 rows x 32 keys per wave) : 4 WMMA
    f32x8 s0 = {0.f, 0.f, 0.f, 0.f, 0.f, 0.f, 0.f, 0.f};
    f32x8 s1 = {0.f, 0.f, 0.f, 0.f, 0.f, 0.f, 0.f, 0.f};
#pragma unroll
    for (int j = 0; j < 2; ++j) {
      const __bf16* b0 = &sK[ln * LK + j * 32];
      const __bf16* b1 = &sK[(16 + ln) * LK + j * 32];
      bf16x16 kf0 = ldfrag(b0 + 8 * hl, b0 + 16 + 8 * hl);
      bf16x16 kf1 = ldfrag(b1 + 8 * hl, b1 + 16 + 8 * hl);
      s0 = wmma_bf16(qf[j], kf0, s0);
      s1 = wmma_bf16(qf[j], kf1, s1);
    }

    // causal mask + online softmax; row reductions via ds_swizzle butterflies
    const int sg0 = sb + ln, sg1 = sb + 16 + ln;
#pragma unroll
    for (int r = 0; r < 8; ++r) {
      const int mr = mbase + r;                  // this element's query row
      const float a  = (sg0 > mr) ? -3.4e38f : (float)s0[r];
      const float c2 = (sg1 > mr) ? -3.4e38f : (float)s1[r];
      float mx = fmaxf(a, c2);
      mx = fmaxf(mx, SWZF(mx, 8));
      mx = fmaxf(mx, SWZF(mx, 4));
      mx = fmaxf(mx, SWZF(mx, 2));
      mx = fmaxf(mx, SWZF(mx, 1));
      const float mn  = fmaxf(mrun[r], mx);      // uniform within 16-lane group
      const float fac = __expf(mrun[r] - mn);
      const float p0 = __expf(a - mn), p1 = __expf(c2 - mn);
      __bf16* prow = &sP[(w * 16 + r + 8 * hl) * LP];
      prow[ln]      = (__bf16)p0;
      prow[16 + ln] = (__bf16)p1;
      lpart[r] = lpart[r] * fac + p0 + p1;       // deferred row-sum reduction
      mrun[r] = mn;
      o[0][r] *= fac; o[1][r] *= fac; o[2][r] *= fac; o[3][r] *= fac;
    }
    lds_wave_fence();                            // sP is wave-private: DScnt only

    // O += P(16x32) * V(32x64) : 4 WMMA
    const __bf16* pbase = &sP[(w * 16 + ln) * LP];
    bf16x16 pf = ldfrag(pbase + 8 * hl, pbase + 16 + 8 * hl);
#pragma unroll
    for (int nt = 0; nt < 4; ++nt) {
      const __bf16* vb = &sVt[(nt * 16 + ln) * LV];
      bf16x16 vf = ldfrag(vb + 8 * hl, vb + 16 + 8 * hl);
      o[nt] = wmma_bf16(pf, vf, o[nt]);
    }
  }

  // finalize row sums (single butterfly), normalize, store bf16 [MR, QN]
  float rinv[8];
#pragma unroll
  for (int r = 0; r < 8; ++r) {
    float l = lpart[r];
    l += SWZF(l, 8);
    l += SWZF(l, 4);
    l += SWZF(l, 2);
    l += SWZF(l, 1);
    rinv[r] = 1.0f / l;
  }
#pragma unroll
  for (int nt = 0; nt < 4; ++nt)
#pragma unroll
    for (int r = 0; r < 8; ++r) {
      const int row = mbase + r;
      const float val = o[nt][r] * rinv[r];
      Om[(size_t)(b * SEQ + row) * QN + h * HD + nt * 16 + ln] = (__bf16)val;
    }
}

// -------------------------------------------------------------------- launcher
extern "C" void kernel_launch(void* const* d_in, const int* in_sizes, int n_in,
                              void* d_out, int out_size, void* d_ws, size_t ws_size,
                              hipStream_t stream) {
  (void)in_sizes; (void)n_in; (void)out_size; (void)ws_size;
  const float* x   = (const float*)d_in[0];
  const float* Wq  = (const float*)d_in[1];
  const float* Wkv = (const float*)d_in[2];
  const float* Wo  = (const float*)d_in[3];
  float* out = (float*)d_out;

  char* ws = (char*)d_ws;
  size_t off = 0;
  __bf16* xb   = (__bf16*)(ws + off); off += (size_t)MR * CH * 2;    // 16 MiB
  __bf16* Wqb  = (__bf16*)(ws + off); off += (size_t)QN * CH * 2;    //  8 MiB
  __bf16* Wkvb = (__bf16*)(ws + off); off += (size_t)KVN * CH * 2;   //  4 MiB
  __bf16* Wob  = (__bf16*)(ws + off); off += (size_t)CH * QN * 2;    //  8 MiB
  __bf16* Qb   = (__bf16*)(ws + off); off += (size_t)MR * QN * 2;    // 16 MiB
  __bf16* KVb  = (__bf16*)(ws + off); off += (size_t)MR * KVN * 2;   //  8 MiB
  __bf16* AOb  = (__bf16*)(ws + off); off += (size_t)MR * QN * 2;    // 16 MiB

  // fp32 -> bf16 casts
  {
    int n = MR * CH;
    cvt_f32_to_bf16<<<n / 1024, 256, 0, stream>>>(x, xb, n / 4);
    n = QN * CH;
    cvt_f32_to_bf16<<<n / 1024, 256, 0, stream>>>(Wq, Wqb, n / 4);
    n = KVN * CH;
    cvt_f32_to_bf16<<<n / 1024, 256, 0, stream>>>(Wkv, Wkvb, n / 4);
    n = CH * QN;
    cvt_f32_to_bf16<<<n / 1024, 256, 0, stream>>>(Wo, Wob, n / 4);
  }

  // Q projection carries the 1/sqrt(D) softmax scale; KV unscaled
  gemm_xwT<true><<<dim3(QN / 128, MR / 128), 256, 0, stream>>>(
      xb, Wqb, Qb, CH, QN, 0.125f);
  gemm_xwT<true><<<dim3(KVN / 128, MR / 128), 256, 0, stream>>>(
      xb, Wkvb, KVb, CH, KVN, 1.0f);

  // causal GQA flash attention
  flash_attn<<<dim3(SEQ / 64, BATCH * NH), 128, 0, stream>>>(Qb, KVb, AOb);

  // output projection (fp32 out)
  gemm_xwT<false><<<dim3(CH / 128, MR / 128), 256, 0, stream>>>(
      AOb, Wob, out, QN, CH, 1.0f);
}
